// DeepFM_61667140436681
// MI455X (gfx1250) — matrix operand order repaired and checked by reading, not compiled, not run on hardware
//
#include <hip/hip_runtime.h>
#include <hip/hip_bf16.h>

typedef _Float16 v8h  __attribute__((ext_vector_type(8)));
typedef _Float16 v16h __attribute__((ext_vector_type(16)));
typedef float    v8f  __attribute__((ext_vector_type(8)));

#define NSAMP 16384
#define NF    39
#define NV    100000
#define NE    16
#define DD    624
#define NPAD  640
#define NL    5

__device__ __forceinline__ v16h cat16(v8h lo, v8h hi) {
    return __builtin_shufflevector(lo, hi, 0,1,2,3,4,5,6,7,8,9,10,11,12,13,14,15);
}

// ---------------------------------------------------------------------------
// Weight prep: W[l] (624 x 624, row=k, col=nout) -> f16 transposed + padded
// WT[mat][nout][k], 640x640, zero padded.  20 matrices: (l*4 + {q,k,v,o}).
// ---------------------------------------------------------------------------
__global__ __launch_bounds__(256)
void prep_weights(const float* __restrict__ Wq, const float* __restrict__ Wk,
                  const float* __restrict__ Wv, const float* __restrict__ Wo,
                  _Float16* __restrict__ WT)
{
    int tid = blockIdx.x * 256 + threadIdx.x;              // 20*640*640 total
    int mat = tid / (NPAD * NPAD);
    int rem = tid - mat * (NPAD * NPAD);
    int nc  = rem / NPAD;          // out-column
    int kk  = rem - nc * NPAD;     // k index
    int l = mat >> 2, w = mat & 3;
    const float* src = (w == 0 ? Wq : w == 1 ? Wk : w == 2 ? Wv : Wo)
                       + (size_t)l * DD * DD;
    float v = 0.0f;
    if (nc < DD && kk < DD) v = src[(size_t)kk * DD + nc];
    WT[(size_t)tid] = (_Float16)v;
}

// ---------------------------------------------------------------------------
// Embedding gather + FM: one block per sample.
// Produces f1 (N x 39 f32), f2 (N x 16 f32), xf16/outf16 (N x 640 f16, padded)
// ---------------------------------------------------------------------------
__global__ __launch_bounds__(128)
void embed_kernel(const int* __restrict__ Xi, const float* __restrict__ Xv,
                  const float* __restrict__ emb1, const float* __restrict__ emb2,
                  _Float16* __restrict__ xf16, _Float16* __restrict__ outf16,
                  float* __restrict__ f1, float* __restrict__ f2)
{
    __shared__ float s_xv[DD];
    __shared__ float s_sum[NE];
    __shared__ float s_ssq[NE];
    __shared__ float s_inv[NE];
    const int n = blockIdx.x;
    const int t = threadIdx.x;
    if (t < NE) { s_sum[t] = 0.0f; s_ssq[t] = 0.0f; }
    __syncthreads();
    for (int p = t; p < DD; p += 128) {
        int f = p >> 4, e = p & 15;
        int idx = Xi[n * NF + f];
        float xvf = Xv[n * NF + f];
        float val = emb2[((size_t)f * NV + idx) * NE + e] * xvf;
        s_xv[p] = val;
        atomicAdd(&s_sum[e], val);
        atomicAdd(&s_ssq[e], val * val);
    }
    if (t < NF) {
        int idx = Xi[n * NF + t];
        f1[(size_t)n * NF + t] = emb1[(size_t)t * NV + idx] * Xv[n * NF + t];
    }
    __syncthreads();
    if (t < NE) {
        float ss  = s_ssq[t];
        float nrm = fmaxf(sqrtf(ss), 1e-12f);
        float inv = 1.0f / nrm;
        float s1  = s_sum[t] * inv; s1 *= s1;
        float s2  = ss * inv * inv;
        f2[(size_t)n * NE + t] = 0.5f * (s1 - s2);
        s_inv[t] = inv;
    }
    __syncthreads();
    for (int p = t; p < NPAD; p += 128) {
        float x = 0.0f;
        if (p < DD) x = s_xv[p] * s_inv[p & 15];
        _Float16 h = (_Float16)x;
        xf16[(size_t)n * NPAD + p]   = h;
        outf16[(size_t)n * NPAD + p] = h;
    }
}

// ---------------------------------------------------------------------------
// WMMA GEMM:  C = A (M x 640 f16) @ Bt^T (+bias).  Bt is [640 nout][640 k] f16.
// Each wave: 32(M) x 64(N) -> 8 accumulators, 8 WMMAs / 12 b128 loads per K.
// Block: 8 waves = 256(M) x 64(N).
// C32 (stride 624, cols<624) and/or C16 (stride 640, all cols) outputs.
// ---------------------------------------------------------------------------
__device__ __forceinline__ void epilogue(v8f acc, int col, int row0,
                                         const float* __restrict__ bias,
                                         float* __restrict__ C32,
                                         _Float16* __restrict__ C16)
{
    float bv = (col < DD) ? bias[col] : 0.0f;
    #pragma unroll
    for (int i = 0; i < 8; ++i) {
        float val = acc[i] + bv;
        if (C32 && col < DD) C32[(size_t)(row0 + i) * DD + col] = val;
        if (C16)             C16[(size_t)(row0 + i) * NPAD + col] = (_Float16)val;
    }
}

__global__ __launch_bounds__(256)
void gemm_wmma(const _Float16* __restrict__ A, const _Float16* __restrict__ Bt,
               const float* __restrict__ bias,
               float* __restrict__ C32, _Float16* __restrict__ C16)
{
    const int lane = threadIdx.x & 31;
    const int wave = threadIdx.x >> 5;
    const int l16  = lane & 15;
    const int hs   = lane >> 4;                 // half-wave select
    const int mBase = blockIdx.x * 256 + wave * 32;
    const int nBase = blockIdx.y * 64;

    // A fragment: lanes 0-15 hold K 0-15; lanes 16-31 hold K 16-31
    const _Float16* aP0 = A + (size_t)(mBase +  0 + l16) * NPAD + (hs ? 16 : 0);
    const _Float16* aP1 = A + (size_t)(mBase + 16 + l16) * NPAD + (hs ? 16 : 0);
    // B fragment: lanes 0-15 hold K 0-15; lanes 16-31 hold K 16-31 (contiguous in Bt)
    const _Float16* bP0 = Bt + (size_t)(nBase +  0 + l16) * NPAD + (hs ? 16 : 0);
    const _Float16* bP1 = Bt + (size_t)(nBase + 16 + l16) * NPAD + (hs ? 16 : 0);
    const _Float16* bP2 = Bt + (size_t)(nBase + 32 + l16) * NPAD + (hs ? 16 : 0);
    const _Float16* bP3 = Bt + (size_t)(nBase + 48 + l16) * NPAD + (hs ? 16 : 0);

    v8f acc00 = {}, acc01 = {}, acc02 = {}, acc03 = {};
    v8f acc10 = {}, acc11 = {}, acc12 = {}, acc13 = {};

    for (int kk = 0; kk < NPAD; kk += 32) {
        v16h af0 = cat16(*(const v8h*)(aP0 + kk), *(const v8h*)(aP0 + kk + 8));
        v16h af1 = cat16(*(const v8h*)(aP1 + kk), *(const v8h*)(aP1 + kk + 8));
        v16h bf0 = cat16(*(const v8h*)(bP0 + kk), *(const v8h*)(bP0 + kk + 8));
        v16h bf1 = cat16(*(const v8h*)(bP1 + kk), *(const v8h*)(bP1 + kk + 8));
        v16h bf2 = cat16(*(const v8h*)(bP2 + kk), *(const v8h*)(bP2 + kk + 8));
        v16h bf3 = cat16(*(const v8h*)(bP3 + kk), *(const v8h*)(bP3 + kk + 8));
        acc00 = __builtin_amdgcn_wmma_f32_16x16x32_f16(false, af0, false, bf0, (short)0, acc00, false, false);
        acc10 = __builtin_amdgcn_wmma_f32_16x16x32_f16(false, af1, false, bf0, (short)0, acc10, false, false);
        acc01 = __builtin_amdgcn_wmma_f32_16x16x32_f16(false, af0, false, bf1, (short)0, acc01, false, false);
        acc11 = __builtin_amdgcn_wmma_f32_16x16x32_f16(false, af1, false, bf1, (short)0, acc11, false, false);
        acc02 = __builtin_amdgcn_wmma_f32_16x16x32_f16(false, af0, false, bf2, (short)0, acc02, false, false);
        acc12 = __builtin_amdgcn_wmma_f32_16x16x32_f16(false, af1, false, bf2, (short)0, acc12, false, false);
        acc03 = __builtin_amdgcn_wmma_f32_16x16x32_f16(false, af0, false, bf3, (short)0, acc03, false, false);
        acc13 = __builtin_amdgcn_wmma_f32_16x16x32_f16(false, af1, false, bf3, (short)0, acc13, false, false);
    }
    const int row0 = mBase + hs * 8;
    const int row1 = mBase + 16 + hs * 8;
    epilogue(acc00, nBase +  0 + l16, row0, bias, C32, C16);
    epilogue(acc01, nBase + 16 + l16, row0, bias, C32, C16);
    epilogue(acc02, nBase + 32 + l16, row0, bias, C32, C16);
    epilogue(acc03, nBase + 48 + l16, row0, bias, C32, C16);
    epilogue(acc10, nBase +  0 + l16, row1, bias, C32, C16);
    epilogue(acc11, nBase + 16 + l16, row1, bias, C32, C16);
    epilogue(acc12, nBase + 32 + l16, row1, bias, C32, C16);
    epilogue(acc13, nBase + 48 + l16, row1, bias, C32, C16);
}

// ---------------------------------------------------------------------------
// Attention elementwise: score[n,h] = 0.25 * <q,k>; att = score * v  (f16 padded)
// ---------------------------------------------------------------------------
__global__ __launch_bounds__(256)
void att_kernel(const float* __restrict__ q, const float* __restrict__ k,
                const float* __restrict__ v, _Float16* __restrict__ att16)
{
    int tid = blockIdx.x * blockDim.x + threadIdx.x;
    if (tid >= NSAMP * NF) return;
    int n = tid / NF, h = tid - n * NF;
    size_t base = (size_t)n * DD + h * 16;
    float s = 0.0f;
    #pragma unroll
    for (int i = 0; i < 16; ++i) s += q[base + i] * k[base + i];
    s *= 0.25f;  // 1/sqrt(DK), DK=16
    size_t ob = (size_t)n * NPAD + h * 16;
    #pragma unroll
    for (int i = 0; i < 16; ++i) att16[ob + i] = (_Float16)(s * v[base + i]);
    if (h < 16) att16[(size_t)n * NPAD + DD + h] = (_Float16)0.0f;  // K padding
}

// ---------------------------------------------------------------------------
// Tail MLP: one wave per sample, shfl_xor reduction.
// ---------------------------------------------------------------------------
__global__ __launch_bounds__(256)
void final_kernel(const float* __restrict__ f1, const float* __restrict__ f2,
                  const float* __restrict__ out32,
                  const float* __restrict__ m1w, const float* __restrict__ m1b,
                  const float* __restrict__ m2w, const float* __restrict__ m2b,
                  const float* __restrict__ m3w, const float* __restrict__ m3b,
                  const float* __restrict__ ffww, const float* __restrict__ ffwb,
                  const float* __restrict__ fdw, const float* __restrict__ fdb,
                  float* __restrict__ outp)
{
    const int lane = threadIdx.x & 31;
    const int wave = threadIdx.x >> 5;
    const int n = blockIdx.x * 8 + wave;

    float h[12];
    #pragma unroll
    for (int j = 0; j < 12; ++j) h[j] = 0.0f;

    for (int f = lane; f < NF; f += 32) {
        float a = f1[(size_t)n * NF + f];
        #pragma unroll
        for (int j = 0; j < 4; ++j) h[j] += a * m1w[f * 4 + j];
    }
    if (lane < NE) {
        float a = f2[(size_t)n * NE + lane];
        #pragma unroll
        for (int j = 0; j < 4; ++j) h[4 + j] += a * m2w[lane * 4 + j];
    }
    for (int d = lane; d < DD; d += 32) {
        float a = out32[(size_t)n * DD + d];
        #pragma unroll
        for (int j = 0; j < 4; ++j) h[8 + j] += a * m3w[d * 4 + j];
    }
    #pragma unroll
    for (int off = 16; off > 0; off >>= 1) {
        #pragma unroll
        for (int j = 0; j < 12; ++j) h[j] += __shfl_xor(h[j], off, 32);
    }
    if (lane == 0) {
        float hb[12];
        #pragma unroll
        for (int j = 0; j < 4; ++j) { hb[j] = h[j] + m1b[j]; hb[4+j] = h[4+j] + m2b[j]; hb[8+j] = h[8+j] + m3b[j]; }
        float fo[12];
        #pragma unroll
        for (int j = 0; j < 12; ++j) {
            float s = ffwb[j];
            #pragma unroll
            for (int i = 0; i < 12; ++i) s += hb[i] * ffww[i * 12 + j];
            fo[j] = fmaxf(s, 0.0f);
        }
        #pragma unroll
        for (int c = 0; c < 2; ++c) {
            float s = fdb[c];
            #pragma unroll
            for (int j = 0; j < 12; ++j) s += fo[j] * fdw[j * 2 + c];
            outp[(size_t)n * 2 + c] = s;
        }
    }
}

// ---------------------------------------------------------------------------
extern "C" void kernel_launch(void* const* d_in, const int* in_sizes, int n_in,
                              void* d_out, int out_size, void* d_ws, size_t ws_size,
                              hipStream_t stream)
{
    const int*   Xi   = (const int*)d_in[0];
    const float* Xv   = (const float*)d_in[1];
    const float* emb1 = (const float*)d_in[2];
    const float* emb2 = (const float*)d_in[3];
    const float* Wq   = (const float*)d_in[4];
    const float* bq   = (const float*)d_in[5];
    const float* Wk   = (const float*)d_in[6];
    const float* bk   = (const float*)d_in[7];
    const float* Wv   = (const float*)d_in[8];
    const float* bv   = (const float*)d_in[9];
    const float* Wo   = (const float*)d_in[10];
    const float* bo   = (const float*)d_in[11];
    const float* m1w  = (const float*)d_in[12];
    const float* m1b  = (const float*)d_in[13];
    const float* m2w  = (const float*)d_in[14];
    const float* m2b  = (const float*)d_in[15];
    const float* m3w  = (const float*)d_in[16];
    const float* m3b  = (const float*)d_in[17];
    const float* ffww = (const float*)d_in[18];
    const float* ffwb = (const float*)d_in[19];
    const float* fdw  = (const float*)d_in[20];
    const float* fdb  = (const float*)d_in[21];

    char* p = (char*)d_ws;
    _Float16* WT     = (_Float16*)p; p += (size_t)20 * NPAD * NPAD * 2;
    _Float16* xf16   = (_Float16*)p; p += (size_t)NSAMP * NPAD * 2;
    _Float16* outf16 = (_Float16*)p; p += (size_t)NSAMP * NPAD * 2;
    _Float16* att16  = (_Float16*)p; p += (size_t)NSAMP * NPAD * 2;
    float*    q32    = (float*)p;    p += (size_t)NSAMP * DD * 4;
    float*    k32    = (float*)p;    p += (size_t)NSAMP * DD * 4;
    float*    v32    = (float*)p;    p += (size_t)NSAMP * DD * 4;
    float*    f1     = (float*)p;    p += (size_t)NSAMP * NF * 4;
    float*    f2     = (float*)p;    p += (size_t)NSAMP * NE * 4;
    float*    out32  = q32;  // q is dead once att16 is built; out reuses it

    prep_weights<<<(20 * NPAD * NPAD) / 256, 256, 0, stream>>>(Wq, Wk, Wv, Wo, WT);
    embed_kernel<<<NSAMP, 128, 0, stream>>>(Xi, Xv, emb1, emb2, xf16, outf16, f1, f2);

    const size_t WSZ = (size_t)NPAD * NPAD;
    dim3 ggrid(NSAMP / 256, NPAD / 64);
    for (int l = 0; l < NL; ++l) {
        gemm_wmma<<<ggrid, 256, 0, stream>>>(xf16,   WT + (l*4 + 0) * WSZ, bq + l * DD, q32, nullptr);
        gemm_wmma<<<ggrid, 256, 0, stream>>>(xf16,   WT + (l*4 + 1) * WSZ, bk + l * DD, k32, nullptr);
        gemm_wmma<<<ggrid, 256, 0, stream>>>(outf16, WT + (l*4 + 2) * WSZ, bv + l * DD, v32, nullptr);
        att_kernel<<<(NSAMP * NF + 255) / 256, 256, 0, stream>>>(q32, k32, v32, att16);
        gemm_wmma<<<ggrid, 256, 0, stream>>>(att16,  WT + (l*4 + 3) * WSZ, bo + l * DD, out32, outf16);
    }

    final_kernel<<<NSAMP / 8, 256, 0, stream>>>(f1, f2, out32, m1w, m1b, m2w, m2b,
                                                m3w, m3b, ffww, ffwb, fdw, fdb,
                                                (float*)d_out);
}
